// CILRSModel_67095979098321
// MI455X (gfx1250) — compile-verified, matrix operand order checked by reading, not run on hardware
//
#include <hip/hip_runtime.h>

#define B_TOT 65536
#define DDIM  512
#define HDIM  256
#define NBR   6
#define TILE_M 128
#define LDE (DDIM + 8)   // padded LDS row stride (elements)
#define LDH (HDIM + 8)

typedef __attribute__((ext_vector_type(16))) __bf16 v16bf;
typedef __attribute__((ext_vector_type(8)))  __bf16 v8bf;
typedef __attribute__((ext_vector_type(4)))  __bf16 v4bf;
typedef __attribute__((ext_vector_type(8)))  float  v8f;

// ---------------------------------------------------------------------------
// Load one 16-lane bf16 fragment row: 8 elems at p, 8 elems at p+16.
// Matches CDNA5 16-bit A/B VGPR layout: lane(0-15) -> row/col = lane&15,
// kb = (lane>>4)*8 selects K chunk {0..7,16..23} vs {8..15,24..31}.
// ---------------------------------------------------------------------------
__device__ __forceinline__ v16bf ldfrag(const __bf16* p) {
    v8bf lo = *reinterpret_cast<const v8bf*>(p);
    v8bf hi = *reinterpret_cast<const v8bf*>(p + 16);
    return __builtin_shufflevector(lo, hi,
        0,1,2,3,4,5,6,7,8,9,10,11,12,13,14,15);
}

// ---------------------------------------------------------------------------
// One WMMA layer with 1xNBLK register blocking: each A fragment feeds NBLK
// WMMAs (4x less LDS traffic per matrix op).
// dst[16-row strip][Nout] = relu(A[16,K] * W^T + bias), bf16 out.
// W stored transposed: [Nout][K] bf16, row-contiguous.
// ---------------------------------------------------------------------------
template <int NBLK>
__device__ __forceinline__ void layer_relu_bf16(
    const __bf16* __restrict__ Asrc, int lda, int K,
    const __bf16* __restrict__ W, const float* __restrict__ bias,
    __bf16* __restrict__ dst, int ldd,
    int m0, int rc, int kb, int mo, int Nout)
{
    const __bf16* Ab = Asrc + (m0 + rc) * lda;
    for (int n0 = 0; n0 < Nout; n0 += 16 * NBLK) {
        const __bf16* Bb[NBLK];
        v8f acc[NBLK];
        #pragma unroll
        for (int j = 0; j < NBLK; ++j) {
            Bb[j]  = W + (n0 + j * 16 + rc) * K;
            acc[j] = (v8f){};
            __builtin_prefetch(Bb[j], 0, 1);     // global_prefetch_b8
        }
        for (int k0 = 0; k0 < K; k0 += 32) {
            v16bf a = ldfrag(Ab + k0 + kb);
            #pragma unroll
            for (int j = 0; j < NBLK; ++j) {
                v16bf b = ldfrag(Bb[j] + k0 + kb);
                acc[j] = __builtin_amdgcn_wmma_f32_16x16x32_bf16(
                             false, a, false, b, (short)0, acc[j], false, false);
            }
        }
        #pragma unroll
        for (int j = 0; j < NBLK; ++j) {
            const int n = n0 + j * 16 + rc;
            const float bv = bias[n];
            #pragma unroll
            for (int v = 0; v < 8; ++v) {
                float x = fmaxf(acc[j][v] + bv, 0.0f);
                dst[(m0 + mo + v) * ldd + n] = (__bf16)x;
            }
        }
    }
}

__global__ __launch_bounds__(256)
void cilrs_fused(const float* __restrict__ embedding,
                 const float* __restrict__ speed,
                 const int*   __restrict__ command,
                 const float* __restrict__ sw1, const float* __restrict__ sb1,
                 const float* __restrict__ sb2,
                 const __bf16* __restrict__ sw2T,
                 const __bf16* __restrict__ bw1T, const float* __restrict__ bb1,
                 const __bf16* __restrict__ bw2T, const float* __restrict__ bb2,
                 const float* __restrict__ bw3,   const float* __restrict__ bb3,
                 const __bf16* __restrict__ ow1T, const float* __restrict__ ob1,
                 const __bf16* __restrict__ ow2T, const float* __restrict__ ob2,
                 const float* __restrict__ ow3,   const float* __restrict__ ob3,
                 float* __restrict__ out_ctrl,    float* __restrict__ out_speed)
{
    extern __shared__ char smem[];
    __bf16* EMB  = reinterpret_cast<__bf16*>(smem);   // [TILE_M][LDE]
    __bf16* BUF1 = EMB  + TILE_M * LDE;               // [TILE_M][LDH] SH / H1 / t1
    __bf16* BUF2 = BUF1 + TILE_M * LDH;               // [TILE_M][LDH] H2 / t2

    const int tid  = threadIdx.x;
    const int lane = tid & 31;
    const int wave = tid >> 5;
    const int m0   = wave * 16;                       // this wave's 16-row strip
    const long row0 = (long)blockIdx.x * TILE_M;

    // ---- Stage A (cooperative): embedding -> bf16 LDS; SH = relu(s*sw1+sb1)
    for (int i = tid; i < TILE_M * (DDIM / 4); i += blockDim.x) {
        int r  = i / (DDIM / 4);
        int c4 = (i - r * (DDIM / 4)) * 4;
        const float4 e = *reinterpret_cast<const float4*>(
            embedding + (row0 + r) * DDIM + c4);
        v4bf v;
        v[0] = (__bf16)e.x; v[1] = (__bf16)e.y;
        v[2] = (__bf16)e.z; v[3] = (__bf16)e.w;
        *reinterpret_cast<v4bf*>(EMB + r * LDE + c4) = v;
    }
    for (int i = tid; i < TILE_M * HDIM; i += blockDim.x) {
        int r = i >> 8;                 // / HDIM
        int h = i & (HDIM - 1);
        float s = speed[row0 + r];
        float x = fmaxf(s * sw1[h] + sb1[h], 0.0f);
        BUF1[r * LDH + h] = (__bf16)x;
    }
    __syncthreads();   // only barrier: after this each wave owns its rows

    const int rc = lane & 15;
    const int kb = (lane >> 4) << 3;    // K sub-chunk base
    const int mo = (lane >> 4) << 3;    // C-fragment M offset for hi lanes

    // ---- GEMM1: EMB(m,n) += SH @ sw2T + sb2  (RMW in LDS), 1x4 blocked ----
    {
        const __bf16* Ab = BUF1 + (m0 + rc) * LDH;
        for (int n0 = 0; n0 < DDIM; n0 += 64) {
            const __bf16* Bb[4];
            v8f acc[4];
            #pragma unroll
            for (int j = 0; j < 4; ++j) {
                Bb[j]  = sw2T + (n0 + j * 16 + rc) * HDIM;
                acc[j] = (v8f){};
                __builtin_prefetch(Bb[j], 0, 1);
            }
            for (int k0 = 0; k0 < HDIM; k0 += 32) {
                v16bf a = ldfrag(Ab + k0 + kb);
                #pragma unroll
                for (int j = 0; j < 4; ++j) {
                    v16bf b = ldfrag(Bb[j] + k0 + kb);
                    acc[j] = __builtin_amdgcn_wmma_f32_16x16x32_bf16(
                                 false, a, false, b, (short)0, acc[j], false, false);
                }
            }
            #pragma unroll
            for (int j = 0; j < 4; ++j) {
                const int n = n0 + j * 16 + rc;
                const float bv = sb2[n];
                #pragma unroll
                for (int v = 0; v < 8; ++v) {
                    __bf16* p = EMB + (m0 + mo + v) * LDE + n;
                    float x = acc[j][v] + bv + (float)(*p);
                    *p = (__bf16)x;
                }
            }
        }
    }

    // ---- 6 branch MLPs (dense per tile, masked write) ----
    for (int br = 0; br < NBR; ++br) {
        layer_relu_bf16<4>(EMB,  LDE, DDIM, bw1T + (long)br * HDIM * DDIM,
                           bb1 + br * HDIM, BUF1, LDH, m0, rc, kb, mo, HDIM);
        layer_relu_bf16<4>(BUF1, LDH, HDIM, bw2T + (long)br * HDIM * HDIM,
                           bb2 + br * HDIM, BUF2, LDH, m0, rc, kb, mo, HDIM);

        // layer 3: [H]->3, split K across lane halves, shfl-reduce
        const int m  = m0 + rc;
        const int kh = (lane >> 4) * (HDIM / 2);
        const float* w3 = bw3 + br * (HDIM * 3);
        const __bf16* hrow = BUF2 + m * LDH;
        float a0 = 0.f, a1 = 0.f, a2 = 0.f;
        for (int k = kh; k < kh + HDIM / 2; ++k) {
            float hv = (float)hrow[k];
            a0 += hv * w3[k * 3 + 0];
            a1 += hv * w3[k * 3 + 1];
            a2 += hv * w3[k * 3 + 2];
        }
        a0 += __shfl_down(a0, 16, 32);
        a1 += __shfl_down(a1, 16, 32);
        a2 += __shfl_down(a2, 16, 32);
        if (lane < 16) {
            long gm = row0 + m;
            if (command[gm] - 1 == br) {
                out_ctrl[gm * 3 + 0] = 1.0f / (1.0f + __expf(-(a0 + bb3[br * 3 + 0])));
                out_ctrl[gm * 3 + 1] = 1.0f / (1.0f + __expf(-(a1 + bb3[br * 3 + 1])));
                out_ctrl[gm * 3 + 2] = 1.0f / (1.0f + __expf(-(a2 + bb3[br * 3 + 2])));
            }
        }
    }

    // ---- speed head: D->H->H->1 ----
    layer_relu_bf16<4>(EMB,  LDE, DDIM, ow1T, ob1, BUF1, LDH, m0, rc, kb, mo, HDIM);
    layer_relu_bf16<4>(BUF1, LDH, HDIM, ow2T, ob2, BUF2, LDH, m0, rc, kb, mo, HDIM);
    {
        const int m  = m0 + rc;
        const int kh = (lane >> 4) * (HDIM / 2);
        const __bf16* hrow = BUF2 + m * LDH;
        float a0 = 0.f;
        for (int k = kh; k < kh + HDIM / 2; ++k)
            a0 += (float)hrow[k] * ow3[k];
        a0 += __shfl_down(a0, 16, 32);
        if (lane < 16) {
            long gm = row0 + m;
            out_speed[gm] = a0 + ob3[0];
        }
    }
}

// ---------------------------------------------------------------------------
// Weight prep: fp32 [batch][R(in)][C(out)] -> bf16 transposed [batch][C][R]
// ---------------------------------------------------------------------------
__global__ void convert_transpose(const float* __restrict__ src,
                                  __bf16* __restrict__ dst,
                                  int R, int C, int total)
{
    int i = blockIdx.x * blockDim.x + threadIdx.x;
    if (i >= total) return;
    int mat = i / (R * C);
    int rem = i - mat * (R * C);
    int r = rem / C;
    int c = rem - r * C;
    dst[mat * (R * C) + c * R + r] = (__bf16)src[i];
}

extern "C" void kernel_launch(void* const* d_in, const int* in_sizes, int n_in,
                              void* d_out, int out_size, void* d_ws, size_t ws_size,
                              hipStream_t stream)
{
    const float* embedding = (const float*)d_in[0];
    const float* speed     = (const float*)d_in[1];
    const int*   command   = (const int*)  d_in[2];
    const float* sw1 = (const float*)d_in[3];
    const float* sb1 = (const float*)d_in[4];
    const float* sw2 = (const float*)d_in[5];
    const float* sb2 = (const float*)d_in[6];
    const float* bw1 = (const float*)d_in[7];
    const float* bb1 = (const float*)d_in[8];
    const float* bw2 = (const float*)d_in[9];
    const float* bb2 = (const float*)d_in[10];
    const float* bw3 = (const float*)d_in[11];
    const float* bb3 = (const float*)d_in[12];
    const float* ow1 = (const float*)d_in[13];
    const float* ob1 = (const float*)d_in[14];
    const float* ow2 = (const float*)d_in[15];
    const float* ob2 = (const float*)d_in[16];
    const float* ow3 = (const float*)d_in[17];
    const float* ob3 = (const float*)d_in[18];

    // bf16 transposed weights in workspace (~2.9 MB, L2-resident)
    __bf16* ws   = (__bf16*)d_ws;
    size_t  o    = 0;
    __bf16* sw2T = ws + o; o += (size_t)HDIM * DDIM;
    __bf16* bw1T = ws + o; o += (size_t)NBR * DDIM * HDIM;
    __bf16* bw2T = ws + o; o += (size_t)NBR * HDIM * HDIM;
    __bf16* ow1T = ws + o; o += (size_t)DDIM * HDIM;
    __bf16* ow2T = ws + o; o += (size_t)HDIM * HDIM;

    auto tr = [&](const float* s, __bf16* dt, int R, int C, int batch) {
        int total = batch * R * C;
        convert_transpose<<<(total + 255) / 256, 256, 0, stream>>>(s, dt, R, C, total);
    };
    tr(sw2, sw2T, HDIM, DDIM, 1);     // [256,512] -> [512][256]
    tr(bw1, bw1T, DDIM, HDIM, NBR);   // [6,512,256] -> 6x[256][512]
    tr(bw2, bw2T, HDIM, HDIM, NBR);   // [6,256,256] -> 6x[256][256]
    tr(ow1, ow1T, DDIM, HDIM, 1);     // [512,256] -> [256][512]
    tr(ow2, ow2T, HDIM, HDIM, 1);     // [256,256] -> [256][256]

    float* out_ctrl  = (float*)d_out;
    float* out_speed = out_ctrl + (size_t)B_TOT * 3;

    const size_t shmem = (size_t)TILE_M * LDE * sizeof(__bf16)
                       + 2 * (size_t)TILE_M * LDH * sizeof(__bf16); // 262 KB
    hipFuncSetAttribute(reinterpret_cast<const void*>(&cilrs_fused),
                        hipFuncAttributeMaxDynamicSharedMemorySize, (int)shmem);

    cilrs_fused<<<B_TOT / TILE_M, 256, shmem, stream>>>(
        embedding, speed, command,
        sw1, sb1, sb2, sw2T,
        bw1T, bb1, bw2T, bb2, bw3, bb3,
        ow1T, ob1, ow2T, ob2, ow3, ob3,
        out_ctrl, out_speed);
}
